// Program_57896159150664
// MI455X (gfx1250) — matrix operand order repaired and checked by reading, not compiled, not run on hardware
//
#include <hip/hip_runtime.h>

typedef __attribute__((ext_vector_type(16))) _Float16 v16h;
typedef __attribute__((ext_vector_type(8)))  float    v8f;

#define STEPS 19

__global__ __launch_bounds__(256) void program_interp_kernel(
    const float* __restrict__ x,
    const float* __restrict__ W1, const float* __restrict__ b1,
    const float* __restrict__ W2, const float* __restrict__ b2,
    const float* __restrict__ W3, const float* __restrict__ b3,
    float* __restrict__ out)
{
    // W3 transposed (3 rows x 64) for broadcast-friendly reads.
    __shared__ __align__(16) float lds_w3t[3 * 64];
    // Per-wave private 16x64 f32 staging tile, row stride 68 (=272B, 16B aligned,
    // shifts bank index by 4 per row -> conflict-light ds_load_b128 reads).
    __shared__ __align__(16) float lds_h2[8][16 * 68];

    const int tid  = threadIdx.x;
    const int wave = tid >> 5;
    const int lane = tid & 31;
    const int h    = lane >> 4;   // wave half (WMMA lane-group)
    const int n    = lane & 15;   // state-in-wave / N index

    if (tid < 192) lds_w3t[tid] = W3[(tid & 63) * 3 + (tid >> 6)];
    __syncthreads();

    // ---------- loop-invariant weight fragments (registers) ----------
    // Layer-1 weights for the 32 features this lane owns in A-fragment layout:
    //   chunk c, pair v, elem e  ->  feature j = 32c + 16*(v>=4) + 8h + 2*(v&3) + e
    float w1r0[32], w1r1[32], b1v[32];
    #pragma unroll
    for (int c = 0; c < 2; ++c)
      #pragma unroll
      for (int v = 0; v < 8; ++v)
        #pragma unroll
        for (int e = 0; e < 2; ++e) {
            const int idx = c * 16 + v * 2 + e;
            const int j = 32 * c + ((v >> 2) << 4) + 8 * h + 2 * (v & 3) + e;
            w1r0[idx] = W1[j];
            w1r1[idx] = W1[64 + j];
            b1v[idx]  = b1[j];
        }

    // W2 as WMMA B-fragments (32x16 f16 per chunk/tile):
    //   lane half h holds K = 32c + 16*(v>=4) + 8h + 2*(v&3) + {0,1}; column N = 16t + n.
    v16h bf[2][4];
    #pragma unroll
    for (int c = 0; c < 2; ++c)
      #pragma unroll
      for (int t = 0; t < 4; ++t)
        #pragma unroll
        for (int v = 0; v < 8; ++v) {
            const int k = 32 * c + ((v >> 2) << 4) + 8 * h + 2 * (v & 3);
            const int j = 16 * t + n;
            bf[c][t][2 * v]     = (_Float16)W2[k * 64 + j];
            bf[c][t][2 * v + 1] = (_Float16)W2[(k + 1) * 64 + j];
        }

    float b2v[4];
    #pragma unroll
    for (int t = 0; t < 4; ++t) b2v[t] = b2[16 * t + n];
    const float b3v0 = b3[0], b3v1 = b3[1], b3v2 = b3[2];

    // ---------- program state (registers, persistent across steps) ----------
    const int s0 = (blockIdx.x * 8 + wave) * 16 + n;
    float pos = x[s0 * 6 + 0];
    float fwd = x[s0 * 6 + 1];
    const float x5 = x[s0 * 6 + 5];

    float* const myh2 = &lds_h2[wave][0];
    const float* const hrow = &lds_h2[wave][n * 68 + 32 * h];
    const float* const w3r0 = &lds_w3t[0 * 64 + 32 * h];
    const float* const w3r1 = &lds_w3t[1 * 64 + 32 * h];
    const float* const w3r2 = &lds_w3t[2 * 64 + 32 * h];

    float zf0 = 0.f, zf1 = 0.f, zf2 = 0.f;

    #pragma unroll 1
    for (int it = 0; it < STEPS; ++it) {
        // ---- layer 1: directly into A-fragment layout -------------------
        v16h af[2];
        #pragma unroll
        for (int c = 0; c < 2; ++c)
          #pragma unroll
          for (int q = 0; q < 16; ++q) {
              const int idx = c * 16 + q;
              float hv = fmaf(pos, w1r0[idx], fmaf(fwd, w1r1[idx], b1v[idx]));
              af[c][q] = (_Float16)fmaxf(hv, 0.f);
          }

        // ---- layer 2: 8x v_wmma_f32_16x16x32_f16 (acc seeded with b2) ---
        v8f acc[4];
        #pragma unroll
        for (int t = 0; t < 4; ++t) {
            #pragma unroll
            for (int r = 0; r < 8; ++r) acc[t][r] = b2v[t];
        }
        #pragma unroll
        for (int t = 0; t < 4; ++t) {
            acc[t] = __builtin_amdgcn_wmma_f32_16x16x32_f16(
                false, af[0], false, bf[0][t], (short)0, acc[t], false, false);
            acc[t] = __builtin_amdgcn_wmma_f32_16x16x32_f16(
                false, af[1], false, bf[1][t], (short)0, acc[t], false, false);
        }

        // ---- ReLU + stage h2 through per-wave LDS tile ------------------
        #pragma unroll
        for (int t = 0; t < 4; ++t)
          #pragma unroll
          for (int r = 0; r < 8; ++r) {
              const int m = r + 8 * h;                  // C/D layout row
              myh2[m * 68 + 16 * t + n] = fmaxf(acc[t][r], 0.f);
          }

        // same-wave DS ops are in-order; just drain DScnt before cross-lane reads
        asm volatile("s_wait_dscnt 0" ::: "memory");

        // ---- layer 3: per-lane dot over 32 features of state n ----------
        float z0 = 0.f, z1 = 0.f, z2 = 0.f;
        #pragma unroll
        for (int i = 0; i < 32; i += 4) {
            const float4 hv = *(const float4*)&hrow[i];
            const float4 a0 = *(const float4*)&w3r0[i];
            const float4 a1 = *(const float4*)&w3r1[i];
            const float4 a2 = *(const float4*)&w3r2[i];
            z0 = fmaf(hv.x, a0.x, z0); z0 = fmaf(hv.y, a0.y, z0);
            z0 = fmaf(hv.z, a0.z, z0); z0 = fmaf(hv.w, a0.w, z0);
            z1 = fmaf(hv.x, a1.x, z1); z1 = fmaf(hv.y, a1.y, z1);
            z1 = fmaf(hv.z, a1.z, z1); z1 = fmaf(hv.w, a1.w, z1);
            z2 = fmaf(hv.x, a2.x, z2); z2 = fmaf(hv.y, a2.y, z2);
            z2 = fmaf(hv.z, a2.z, z2); z2 = fmaf(hv.w, a2.w, z2);
        }
        // combine the two 32-feature halves (lanes n <-> n+16)
        z0 += __shfl_xor(z0, 16, 32);
        z1 += __shfl_xor(z1, 16, 32);
        z2 += __shfl_xor(z2, 16, 32);
        z0 += b3v0; z1 += b3v1; z2 += b3v2;

        // argmax on logits (sigmoid is monotonic); jnp.argmax = first max
        float mx = z0; int amax = 0;
        if (z1 > mx) { mx = z1; amax = 1; }
        if (z2 > mx) { amax = 2; }
        pos += (float)(amax - 1);
        fwd += 1.0f;
        zf0 = z0; zf1 = z1; zf2 = z2;
    }

    // ---- epilogue: materialize final state (lanes 0..15 only) ----------
    if (h == 0) {
        const float p0 = 1.0f / (1.0f + __expf(-zf0));
        const float p1 = 1.0f / (1.0f + __expf(-zf1));
        const float p2 = 1.0f / (1.0f + __expf(-zf2));
        float* o = out + (size_t)s0 * 6;
        o[0] = pos;
        o[1] = fwd;                     // x1 + 19
        o[2] = p0; o[3] = p1; o[4] = p2;
        o[5] = x5 + (float)STEPS;       // step counter
    }
}

extern "C" void kernel_launch(void* const* d_in, const int* in_sizes, int n_in,
                              void* d_out, int out_size, void* d_ws, size_t ws_size,
                              hipStream_t stream) {
    (void)n_in; (void)d_ws; (void)ws_size; (void)out_size;
    const float* x  = (const float*)d_in[0];
    const float* W1 = (const float*)d_in[1];
    const float* b1 = (const float*)d_in[2];
    const float* W2 = (const float*)d_in[3];
    const float* b2 = (const float*)d_in[4];
    const float* W3 = (const float*)d_in[5];
    const float* b3 = (const float*)d_in[6];
    float* out = (float*)d_out;

    const int Bn = in_sizes[0] / 6;          // 1048576 states
    const int blocks = Bn / 128;             // 8 waves * 16 states per block
    program_interp_kernel<<<blocks, 256, 0, stream>>>(x, W1, b1, W2, b2, W3, b3, out);
}
